// BiLSTMTagger_52304111730763
// MI455X (gfx1250) — matrix operand order repaired and checked by reading, not compiled, not run on hardware
//
#include <hip/hip_runtime.h>
#include <hip/hip_bf16.h>
#include <math.h>

// ---------------------------------------------------------------------------
// BiLSTM tagger for MI455X (gfx1250, wave32, WMMA).
// All GEMMs run on v_wmma_f32_16x16x32_bf16. LSTM recurrence runs in a
// persistent single-workgroup kernel per direction with h/c state in LDS.
//
// Assumed d_in flattening (setup_inputs order, params dict recursed in
// insertion order):
//  0 sentence(B*T i32) 1 p_sentence 2 pos_tags 3 target_idx_in(B)
//  4 region_marks(B*T f32) 5 local_roles_voc(B*R) 6 frames(B*R)
//  7 local_roles_mask(B*R f32) 8 sent_pred_lemmas_idx 9 dep_tags
// 10 dep_heads 11 targets
// 12 emb_word 13 emb_fixed 14 emb_pos 15 emb_lemma 16 emb_role 17 emb_frame
// 18..21 share0.fw{wih,whh,bih,bhh} 22..25 share0.bw
// 26..29 share1.fw 30..33 share1.bw  34..37 spe.fw 38..41 spe.bw
// 42..45 srl.fw 46..49 srl.bw
// 50 m_w 51 m_b 52 h_w 53 h_b 54 mlp_w 55 mlp_b 56 rm_w 57 rm_b
//
// d_out: [srl_loss, dep_loss, loss, srl_probs(16384*40), dep_tag_space(16384*50)]
// ---------------------------------------------------------------------------

typedef __bf16 bf16_t;
typedef __attribute__((ext_vector_type(16))) __bf16 v16bf;
typedef __attribute__((ext_vector_type(8)))  float  v8f;
typedef __attribute__((ext_vector_type(4)))  float  v4f;

#define BB   64
#define TT   256
#define HH   300
#define HPAD 304     // hidden units padded to 16
#define KH   320     // hidden K padded to 32 for WMMA
#define NG   1216    // 4 * HPAD : permuted gate width
#define LDX  608     // bilstm layer output width (2 * HPAD)
#define RR   40
#define NROWS (BB*TT)

__device__ __forceinline__ v8f wmma_bf16(v16bf a, v16bf b, v8f c) {
  return __builtin_amdgcn_wmma_f32_16x16x32_bf16(false, a, false, b, (short)0, c,
                                                 false, false);
}

// A-matrix fragment (16 x 32, bf16, row-major src, ld in elements).
// lanes 0-15: M=lane, K chunks [0..7],[16..23]; lanes 16-31: K [8..15],[24..31]
__device__ __forceinline__ v16bf load_a_frag(const bf16_t* base, int ld) {
  const int lane = threadIdx.x & 31;
  const int row  = lane & 15;
  const int k0   = (lane >> 4) << 3;
  union { v16bf v; v4f f[2]; } u;
  const bf16_t* p = base + (size_t)row * ld + k0;
  u.f[0] = *(const v4f*)(p);
  u.f[1] = *(const v4f*)(p + 16);
  return u.v;
}

// B-matrix fragment from Bt (N x K row-major == B column-major), unguarded.
// lanes 0-15: N=lane&15, K [k..k+7],[k+16..k+23]; lanes 16-31: K [k+8..k+15],[k+24..k+31]
__device__ __forceinline__ v16bf load_b_frag_fast(const bf16_t* BtRow0, int ld,
                                                  int k) {
  const int lane = threadIdx.x & 31;
  const bf16_t* p = BtRow0 + (size_t)(lane & 15) * ld + k + ((lane >> 4) << 3);
  union { v16bf v; v4f f[2]; } u;
  u.f[0] = *(const v4f*)(p);
  u.f[1] = *(const v4f*)(p + 16);
  return u.v;
}

// Guarded B-fragment (edge tiles only): rows >= maxRow read as zero.
__device__ __forceinline__ v16bf load_b_frag_guard(const bf16_t* Bt, int ld,
                                                   int row0, int maxRow, int k) {
  const int lane = threadIdx.x & 31;
  const int row  = row0 + (lane & 15);
  const int kk   = k + ((lane >> 4) << 3);
  union { v16bf v; v4f f[2]; } u;
  if (row < maxRow) {
    const bf16_t* p = Bt + (size_t)row * ld + kk;
    u.f[0] = *(const v4f*)(p);
    u.f[1] = *(const v4f*)(p + 16);
  } else {
    v4f z = {0.f, 0.f, 0.f, 0.f};
    u.f[0] = z; u.f[1] = z;
  }
  return u.v;
}

// ---------------------------------------------------------------------------
// Generic bf16 WMMA GEMM:  C[MxN] = act(A[MxKp] * Bt[NxKp]^T + bias)
// block = 256 threads = 8 waves; block tile 128(M) x 64(N); wave tile 16x64.
// act: 0 none, 1 tanh, 2 relu.  outmap: 0 identity, 1 n -> n + 4*(n/300).
// Full interior N-tiles take a branch-free unguarded inner loop.
// ---------------------------------------------------------------------------
__global__ void gemm_bf16_kernel(const bf16_t* __restrict__ A,
                                 const bf16_t* __restrict__ Bt,
                                 const float* __restrict__ bias,
                                 float* Cf, int ldcf, bf16_t* Cb, int ldcb,
                                 int M, int N, int Kp, int act, int outmap) {
  const int wave = threadIdx.x >> 5;
  const int lane = threadIdx.x & 31;
  const int n0 = blockIdx.x * 64;
  const int m0 = blockIdx.y * 128 + wave * 16;
  if (m0 >= M) return;
  const int lrow = (lane >> 4) << 3;
  const int lcol = lane & 15;
  const v8f vz = {0.f,0.f,0.f,0.f,0.f,0.f,0.f,0.f};
  v8f acc[4] = {vz, vz, vz, vz};

  const bf16_t* Arow = A + (size_t)m0 * Kp;
  if (n0 + 64 <= N) {               // block-uniform: interior tile, no guards
    const bf16_t* B0 = Bt + (size_t)n0 * Kp;
    for (int k = 0; k < Kp; k += 32) {
      if (k + 32 < Kp) __builtin_prefetch(Arow + (size_t)lcol * Kp + k + 32, 0, 1);
      v16bf a = load_a_frag(Arow + k, Kp);
#pragma unroll
      for (int j = 0; j < 4; ++j) {
        v16bf b = load_b_frag_fast(B0 + (size_t)(j * 16) * Kp, Kp, k);
        acc[j] = wmma_bf16(a, b, acc[j]);
      }
    }
  } else {                          // edge tile: guard B rows against N
    for (int k = 0; k < Kp; k += 32) {
      v16bf a = load_a_frag(Arow + k, Kp);
#pragma unroll
      for (int j = 0; j < 4; ++j) {
        v16bf b = load_b_frag_guard(Bt, Kp, n0 + j * 16, N, k);
        acc[j] = wmma_bf16(a, b, acc[j]);
      }
    }
  }
#pragma unroll
  for (int j = 0; j < 4; ++j) {
    int n = n0 + j * 16 + lcol;
    bool valid = n < N;
    float bi = (bias != nullptr && valid) ? bias[n] : 0.f;
#pragma unroll
    for (int r = 0; r < 8; ++r) {
      int m = m0 + r + lrow;
      float v = acc[j][r] + bi;
      if (act == 1) v = tanhf(v);
      else if (act == 2) v = fmaxf(v, 0.f);
      if (valid && Cf) Cf[(size_t)m * ldcf + n] = v;
      if (Cb) {
        if (valid) {
          int nn = (outmap == 1) ? (n + 4 * (n / 300)) : n;
          Cb[(size_t)m * ldcb + nn] = (bf16_t)v;
        } else if (outmap == 0) {
          Cb[(size_t)m * ldcb + n] = (bf16_t)0.f;   // zero-pad tile columns
        }
      }
    }
  }
}

// ---------------------------------------------------------------------------
// Persistent LSTM scan: one workgroup (512 thr = 16 waves) per direction.
// gates(t) = xg[:,t,:] + h(t-1) @ whh_perm^T   (WMMA, h in LDS as bf16)
// column layout: tile j (19 tiles of 16 units) -> [i|f|g|o] blocks of 16.
// c state lives in LDS in fragment order; h written to LDS + next-layer X.
// All whh rows (NG=1216) are valid -> unguarded B loads.
// ---------------------------------------------------------------------------
#define SCAN_TASKS 76           // 4 m-tiles * 19 u-tiles
#define SCAN_THREADS 512
#define SCAN_LDS (64*KH*2 + SCAN_TASKS*256*4)

__global__ void lstm_scan_kernel(const float* __restrict__ xg,
                                 const bf16_t* __restrict__ whh,
                                 bf16_t* __restrict__ Xout,
                                 int ldo, int dirOff, int reverse) {
  extern __shared__ char smem[];
  bf16_t* hbuf = (bf16_t*)smem;                 // 64 x KH
  float*  cbuf = (float*)(smem + 64 * KH * 2);  // SCAN_TASKS x 256

  for (int i = threadIdx.x; i < 64 * KH; i += blockDim.x) hbuf[i] = (bf16_t)0.f;
  for (int i = threadIdx.x; i < SCAN_TASKS * 256; i += blockDim.x) cbuf[i] = 0.f;
  __syncthreads();

  const int wave = threadIdx.x >> 5;
  const int lane = threadIdx.x & 31;
  const int nw   = blockDim.x >> 5;
  const int lrow = (lane >> 4) << 3;
  const int lcol = lane & 15;
  const v8f vz = {0.f,0.f,0.f,0.f,0.f,0.f,0.f,0.f};
  float hreg[5][8];

  for (int t = 0; t < TT; ++t) {
    const int ta = reverse ? (TT - 1 - t) : t;
    __syncthreads();                       // h(t-1) stable
    int ti = 0;
    for (int task = wave; task < SCAN_TASKS; task += nw, ++ti) {
      const int mtile = task & 3;
      const int utile = task >> 2;
      v8f acc[4] = {vz, vz, vz, vz};
      const bf16_t* hA = hbuf + (size_t)(mtile * 16) * KH;
      const bf16_t* B0 = whh + (size_t)(utile * 64) * KH;
      for (int k = 0; k < KH; k += 32) {
        v16bf a = load_a_frag(hA + k, KH);
#pragma unroll
        for (int g = 0; g < 4; ++g) {
          v16bf b = load_b_frag_fast(B0 + (size_t)(g * 16) * KH, KH, k);
          acc[g] = wmma_bf16(a, b, acc[g]);
        }
      }
      float* cw = cbuf + task * 256;
      const int ubase = utile * 64;
#pragma unroll
      for (int r = 0; r < 8; ++r) {
        const int m = mtile * 16 + r + lrow;           // batch 0..63
        const size_t row = (size_t)m * TT + ta;
        const float* xr = xg + row * NG + ubase;
        float gi = acc[0][r] + xr[lcol];
        float gf = acc[1][r] + xr[16 + lcol];
        float gg = acc[2][r] + xr[32 + lcol];
        float go = acc[3][r] + xr[48 + lcol];
        float c  = cw[r * 32 + lane];
        float si = 1.f / (1.f + __expf(-gi));
        float sf = 1.f / (1.f + __expf(-gf));
        float so = 1.f / (1.f + __expf(-go));
        c = sf * c + si * tanhf(gg);
        cw[r * 32 + lane] = c;
        float h = so * tanhf(c);
        hreg[ti][r] = h;
        Xout[row * ldo + dirOff + utile * 16 + lcol] = (bf16_t)h;
      }
    }
    __syncthreads();                       // all reads of h(t-1) done
    ti = 0;
    for (int task = wave; task < SCAN_TASKS; task += nw, ++ti) {
      const int mtile = task & 3;
      const int utile = task >> 2;
#pragma unroll
      for (int r = 0; r < 8; ++r)
        hbuf[(size_t)(mtile * 16 + r + lrow) * KH + utile * 16 + lcol] =
            (bf16_t)hreg[ti][r];
    }
  }
}

// ---------------------------------------------------------------------------
// Per-batch tag_space GEMM: out[b,t,r] = sum_d h4[b,t,d]*mapped[b,r,d] + maskbias
// h4 = [hs3 (padded-split 608) | pred-row broadcast (608)] read on the fly.
// block = 128 thr (4 waves); grid = (64 batches, 4); wave tile 16(T) x 48(R).
// Role tiles 0,1 (rows 0..31 < 40) are full -> unguarded; tile 2 guarded.
// ---------------------------------------------------------------------------
__global__ void tag_gemm_kernel(const bf16_t* __restrict__ X5,
                                const bf16_t* __restrict__ mapped,
                                const int* __restrict__ tIdx,
                                const float* __restrict__ mask,
                                float* __restrict__ out) {
  const int b    = blockIdx.x;
  const int wave = threadIdx.x >> 5;
  const int lane = threadIdx.x & 31;
  const int mt   = blockIdx.y * 4 + wave;      // t-tile 0..15
  const int lrow = (lane >> 4) << 3;
  const int lcol = lane & 15;
  const int k0a  = lrow;                       // A-frag k offset (0 or 8)
  const v8f vz = {0.f,0.f,0.f,0.f,0.f,0.f,0.f,0.f};
  v8f acc[3] = {vz, vz, vz};

  const int pr = tIdx[b];
  const bf16_t* Xb = X5 + ((size_t)b * TT + mt * 16 + lcol) * LDX;
  const bf16_t* Pr = X5 + ((size_t)b * TT + pr) * LDX;
  const bf16_t* Mb = mapped + (size_t)b * RR * NG;

  for (int k = 0; k < 2 * LDX; k += 32) {
    union { v16bf v; v4f f[2]; } ua;
    const bf16_t* pa = (k < LDX) ? (Xb + k + k0a) : (Pr + (k - LDX) + k0a);
    ua.f[0] = *(const v4f*)(pa);
    ua.f[1] = *(const v4f*)(pa + 16);
    v16bf b0 = load_b_frag_fast(Mb, NG, k);
    acc[0] = wmma_bf16(ua.v, b0, acc[0]);
    v16bf b1 = load_b_frag_fast(Mb + (size_t)16 * NG, NG, k);
    acc[1] = wmma_bf16(ua.v, b1, acc[1]);
    v16bf b2 = load_b_frag_guard(Mb, NG, 32, RR, k);
    acc[2] = wmma_bf16(ua.v, b2, acc[2]);
  }
#pragma unroll
  for (int j = 0; j < 3; ++j) {
    const int rcol = j * 16 + lcol;
    if (rcol >= RR) continue;
    const float mb = (mask[b * RR + rcol] - 1.f) * 1000000.0f;
#pragma unroll
    for (int r = 0; r < 8; ++r) {
      const int m = mt * 16 + r + lrow;
      out[((size_t)b * TT + m) * RR + rcol] = acc[j][r] + mb;
    }
  }
}

// ---------------------------------------------------------------------------
// Builders / converters
// ---------------------------------------------------------------------------
__global__ void embed_kernel(const int* __restrict__ sent, const int* __restrict__ psent,
                             const int* __restrict__ pos,  const int* __restrict__ lemma,
                             const float* __restrict__ region,
                             const float* __restrict__ eW, const float* __restrict__ eF,
                             const float* __restrict__ eP, const float* __restrict__ eL,
                             bf16_t* __restrict__ X1) {
  const int row = blockIdx.x;
  const int ws = sent[row], wp = psent[row], pp = pos[row], lm = lemma[row];
  bf16_t* o = X1 + (size_t)row * 352;
  for (int c = threadIdx.x; c < 352; c += blockDim.x) {
    float v;
    if (c < 100)      v = eW[(size_t)ws * 100 + c];
    else if (c < 200) v = eF[(size_t)wp * 100 + c - 100];
    else if (c < 232) v = eP[(size_t)pp * 32 + c - 200];
    else if (c < 332) v = eL[(size_t)lm * 100 + c - 232];
    else if (c == 332) v = region[row];
    else v = 0.f;
    o[c] = (bf16_t)v;
  }
}

// permuted-gate weight conversion. Out row n: utile=n>>6, gate=(n>>4)&3,
// unit=utile*16+(n&15). kmode 0: identity pad; kmode 1: 600 -> 608 split.
__global__ void conv_wih_kernel(const float* __restrict__ w,
                                const float* __restrict__ bih,
                                const float* __restrict__ bhh,
                                bf16_t* __restrict__ Wout, float* bOut,
                                int inDim, int Kp, int kmode) {
  const int n = blockIdx.x;
  const int utile = n >> 6, gate = (n >> 4) & 3, ul = n & 15;
  const int u = utile * 16 + ul;
  const bool valid = u < HH;
  const int srcRow = gate * HH + u;
  if (bOut && threadIdx.x == 0)
    bOut[n] = valid ? (bih[srcRow] + bhh[srcRow]) : 0.f;
  for (int c = threadIdx.x; c < Kp; c += blockDim.x) {
    int sk;
    if (kmode == 0) sk = (c < inDim) ? c : -1;
    else { int half = c / HPAD, cc = c % HPAD; sk = (cc < HH) ? half * HH + cc : -1; }
    float v = (valid && sk >= 0) ? w[(size_t)srcRow * inDim + sk] : 0.f;
    Wout[(size_t)n * Kp + c] = (bf16_t)v;
  }
}

__global__ void build_a2_kernel(const bf16_t* __restrict__ X3,
                                const int* __restrict__ dep_heads,
                                bf16_t* __restrict__ A2) {
  const int row = blockIdx.x;
  const int b = row >> 8;
  const int head = dep_heads[row];
  const int idx = (head - 1 > 0) ? head - 1 : 0;
  const bf16_t* self = X3 + (size_t)row * LDX;
  const bf16_t* gat  = X3 + ((size_t)(b << 8) + idx) * LDX;
  const bool hm = head > 0;
  for (int c = threadIdx.x; c < 256; c += blockDim.x) {
    bf16_t v;
    if (c < 50)       v = self[c];
    else if (c < 100) v = self[HPAD + c - 50];
    else if (c < 150) v = hm ? gat[c - 100] : (bf16_t)0.f;
    else if (c < 200) v = hm ? gat[HPAD + c - 150] : (bf16_t)0.f;
    else              v = (bf16_t)0.f;
    A2[(size_t)row * 256 + c] = v;
  }
}

__global__ void conv_depw_kernel(const float* __restrict__ mW, const float* __restrict__ hW,
                                 const float* __restrict__ mB, const float* __restrict__ hB,
                                 bf16_t* __restrict__ Wout, float* __restrict__ bOut) {
  const int o = blockIdx.x;   // 0..199
  if (threadIdx.x == 0) bOut[o] = mB[o] + hB[o];
  for (int c = threadIdx.x; c < 256; c += blockDim.x) {
    float v = 0.f;
    if (c < 100)      v = mW[(size_t)o * 100 + c];
    else if (c < 200) v = hW[(size_t)o * 100 + c - 100];
    Wout[(size_t)o * 256 + c] = (bf16_t)v;
  }
}

__global__ void conv_pad_kernel(const float* __restrict__ src, bf16_t* __restrict__ dst,
                                int Kin, int Kp) {
  const int n = blockIdx.x;
  for (int c = threadIdx.x; c < Kp; c += blockDim.x)
    dst[(size_t)n * Kp + c] = (bf16_t)((c < Kin) ? src[(size_t)n * Kin + c] : 0.f);
}

__global__ void build_rolecat_kernel(const float* __restrict__ eR, const float* __restrict__ eFr,
                                     const int* __restrict__ voc, const int* __restrict__ fr,
                                     bf16_t* __restrict__ out) {
  const int row = blockIdx.x;   // b*40+r
  const int rv = voc[row], fv = fr[row];
  bf16_t* o = out + (size_t)row * 256;
  for (int c = threadIdx.x; c < 256; c += blockDim.x) {
    float v = (c < 128) ? eR[(size_t)rv * 128 + c] : eFr[(size_t)fv * 128 + c - 128];
    o[c] = (bf16_t)v;
  }
}

__global__ void fill_f32_kernel(float* p, long long n) {
  long long i = (long long)blockIdx.x * blockDim.x + threadIdx.x;
  if (i < n) p[i] = 0.f;
}
__global__ void fill_bf16_kernel(bf16_t* p, long long n) {
  long long i = (long long)blockIdx.x * blockDim.x + threadIdx.x;
  if (i < n) p[i] = (bf16_t)0.f;
}

// ---------------------------------------------------------------------------
// Softmax + masked CE (ignore label 0). One wave per row.
// accum[0]+=nll, accum[1]+=1 for valid rows.
// ---------------------------------------------------------------------------
__global__ void ce_softmax_kernel(const float* __restrict__ logits, int C, int ld,
                                  const int* __restrict__ targets,
                                  float* probs, float* accum, int Nrows) {
  const int row = blockIdx.x * (blockDim.x >> 5) + (threadIdx.x >> 5);
  if (row >= Nrows) return;
  const int lane = threadIdx.x & 31;
  const float* lr = logits + (size_t)row * ld;
  float mx = -1e30f;
  for (int c = lane; c < C; c += 32) mx = fmaxf(mx, lr[c]);
  for (int o = 16; o; o >>= 1) mx = fmaxf(mx, __shfl_xor(mx, o, 32));
  float s = 0.f;
  for (int c = lane; c < C; c += 32) s += __expf(lr[c] - mx);
  for (int o = 16; o; o >>= 1) s += __shfl_xor(s, o, 32);
  if (probs) {
    float inv = 1.f / s;
    for (int c = lane; c < C; c += 32)
      probs[(size_t)row * C + c] = __expf(lr[c] - mx) * inv;
  }
  if (lane == 0) {
    int t = targets[row];
    if (t != 0) {
      float nll = -(lr[t] - mx - __logf(s));
      atomicAdd(&accum[0], nll);
      atomicAdd(&accum[1], 1.f);
    }
  }
}

__global__ void finalize_kernel(const float* __restrict__ accum, float* __restrict__ out) {
  float srl = accum[0] / fmaxf(accum[1], 1.f);
  float dep = accum[2] / fmaxf(accum[3], 1.f);
  out[0] = srl;
  out[1] = dep;
  out[2] = srl + 0.1f * dep;
}

// ---------------------------------------------------------------------------
extern "C" void kernel_launch(void* const* d_in, const int* in_sizes, int n_in,
                              void* d_out, int out_size, void* d_ws, size_t ws_size,
                              hipStream_t stream) {
  (void)in_sizes; (void)n_in; (void)out_size; (void)ws_size;

  // ---- inputs ----
  const int*   sentence   = (const int*)d_in[0];
  const int*   p_sentence = (const int*)d_in[1];
  const int*   pos_tags   = (const int*)d_in[2];
  const int*   target_idx = (const int*)d_in[3];
  const float* region     = (const float*)d_in[4];
  const int*   roles_voc  = (const int*)d_in[5];
  const int*   frames     = (const int*)d_in[6];
  const float* roles_mask = (const float*)d_in[7];
  const int*   lemmas     = (const int*)d_in[8];
  const int*   dep_tags   = (const int*)d_in[9];
  const int*   dep_heads  = (const int*)d_in[10];
  const int*   targets    = (const int*)d_in[11];
  const float* emb_word   = (const float*)d_in[12];
  const float* emb_fixed  = (const float*)d_in[13];
  const float* emb_pos    = (const float*)d_in[14];
  const float* emb_lemma  = (const float*)d_in[15];
  const float* emb_role   = (const float*)d_in[16];
  const float* emb_frame  = (const float*)d_in[17];
  const float* m_w   = (const float*)d_in[50];
  const float* m_b   = (const float*)d_in[51];
  const float* h_w   = (const float*)d_in[52];
  const float* h_b   = (const float*)d_in[53];
  const float* mlp_w = (const float*)d_in[54];
  const float* mlp_b = (const float*)d_in[55];
  const float* rm_w  = (const float*)d_in[56];
  const float* rm_b  = (const float*)d_in[57];

  // ---- workspace carve ----
  char* cur = (char*)d_ws;
  auto alloc = [&](size_t b) -> char* {
    char* p = cur; cur += (b + 255) & ~(size_t)255; return p;
  };
  bf16_t* X1 = (bf16_t*)alloc((size_t)NROWS * 352 * 2);
  bf16_t* X2 = (bf16_t*)alloc((size_t)NROWS * LDX * 2);
  bf16_t* X3 = (bf16_t*)alloc((size_t)NROWS * LDX * 2);
  bf16_t* X4 = (bf16_t*)alloc((size_t)NROWS * LDX * 2);
  bf16_t* X5 = (bf16_t*)alloc((size_t)NROWS * LDX * 2);
  float*  xg = (float*)alloc((size_t)NROWS * NG * 4);
  bf16_t* wperm   = (bf16_t*)alloc((size_t)NG * LDX * 2);
  bf16_t* whhperm = (bf16_t*)alloc((size_t)NG * KH * 2);
  float*  biasv   = (float*)alloc(NG * 4);
  bf16_t* A2   = (bf16_t*)alloc((size_t)NROWS * 256 * 2);
  bf16_t* W2t  = (bf16_t*)alloc((size_t)200 * 256 * 2);
  float*  bias2 = (float*)alloc(200 * 4);
  bf16_t* D2   = (bf16_t*)alloc((size_t)NROWS * 256 * 2);
  bf16_t* mlpW = (bf16_t*)alloc((size_t)50 * 256 * 2);
  bf16_t* roleCat = (bf16_t*)alloc((size_t)BB * RR * 256 * 2);
  bf16_t* rmW     = (bf16_t*)alloc((size_t)1200 * 256 * 2);
  bf16_t* mappedB = (bf16_t*)alloc((size_t)BB * RR * NG * 2);
  float*  tagLogits = (float*)alloc((size_t)NROWS * RR * 4);
  float*  accum     = (float*)alloc(4 * 4);   // srlSum, srlCnt, depSum, depCnt

  float* outScal  = (float*)d_out;
  float* srlProbs = outScal + 3;
  float* depOut   = srlProbs + (size_t)NROWS * RR;

  // ---- init accumulators ----
  fill_f32_kernel<<<1, 32, 0, stream>>>(accum, 4);

  // ---- embeddings -> X1 (16384 x 352 bf16) ----
  embed_kernel<<<NROWS, 128, 0, stream>>>(sentence, p_sentence, pos_tags, lemmas,
                                          region, emb_word, emb_fixed, emb_pos,
                                          emb_lemma, X1);

  // ---- 4 BiLSTM layers (share0, share1, spe, srl) ----
  bf16_t* Xs[5] = {X1, X2, X3, X4, X5};
  const int wbase[4]  = {18, 26, 34, 42};
  const int Kps[4]    = {352, 608, 608, 608};
  const int inDims[4] = {333, 600, 600, 600};
  const int kmodes[4] = {0, 1, 1, 1};

  for (int l = 0; l < 4; ++l) {
    for (int d = 0; d < 2; ++d) {
      const int base = wbase[l] + 4 * d;
      const float* wih = (const float*)d_in[base + 0];
      const float* whh = (const float*)d_in[base + 1];
      const float* bih = (const float*)d_in[base + 2];
      const float* bhh = (const float*)d_in[base + 3];
      conv_wih_kernel<<<NG, 128, 0, stream>>>(wih, bih, bhh, wperm, biasv,
                                              inDims[l], Kps[l], kmodes[l]);
      conv_wih_kernel<<<NG, 128, 0, stream>>>(whh, nullptr, nullptr, whhperm,
                                              nullptr, HH, KH, 0);
      // xg = X @ wperm^T + bias   (16384 x 1216)
      gemm_bf16_kernel<<<dim3(NG / 64, NROWS / 128), 256, 0, stream>>>(
          Xs[l], wperm, biasv, xg, NG, nullptr, 0, NROWS, NG, Kps[l], 0, 0);
      // recurrent scan -> next X (fw cols 0..303, bw cols 304..607)
      lstm_scan_kernel<<<1, SCAN_THREADS, SCAN_LDS, stream>>>(
          xg, whhperm, Xs[l + 1], LDX, d * HPAD, d);
    }
  }

  // ---- dependency head (from hs1 == X3) ----
  build_a2_kernel<<<NROWS, 128, 0, stream>>>(X3, dep_heads, A2);
  conv_depw_kernel<<<200, 128, 0, stream>>>(m_w, h_w, m_b, h_b, W2t, bias2);
  // dep_hidden = tanh(A2 @ W2t^T + bias2) -> D2 bf16 (zero-padded to 256)
  gemm_bf16_kernel<<<dim3(4, NROWS / 128), 256, 0, stream>>>(
      A2, W2t, bias2, nullptr, 0, D2, 256, NROWS, 200, 256, 1, 0);
  conv_pad_kernel<<<50, 128, 0, stream>>>(mlp_w, mlpW, 200, 256);
  // dep_tag_space = D2 @ mlp_w^T + mlp_b -> d_out (16384 x 50)
  gemm_bf16_kernel<<<dim3(1, NROWS / 128), 256, 0, stream>>>(
      D2, mlpW, mlp_b, depOut, 50, nullptr, 0, NROWS, 50, 256, 0, 0);

  // ---- SRL head ----
  build_rolecat_kernel<<<BB * RR, 128, 0, stream>>>(emb_role, emb_frame,
                                                    roles_voc, frames, roleCat);
  conv_pad_kernel<<<1200, 128, 0, stream>>>(rm_w, rmW, 256, 256);
  {
    long long n = (long long)BB * RR * NG;
    fill_bf16_kernel<<<(unsigned)((n + 255) / 256), 256, 0, stream>>>(mappedB, n);
  }
  // mapped = relu(roleCat @ rm_w^T + rm_b), stored with split304 column remap
  gemm_bf16_kernel<<<dim3(NG / 64, (BB * RR) / 128), 256, 0, stream>>>(
      roleCat, rmW, rm_b, nullptr, 0, mappedB, NG, BB * RR, 1200, 256, 2, 1);
  // tag_space = h4 . mapped + mask bias
  tag_gemm_kernel<<<dim3(BB, 4), 128, 0, stream>>>(X5, mappedB, target_idx,
                                                   roles_mask, tagLogits);

  // ---- losses / softmax outputs ----
  ce_softmax_kernel<<<NROWS / 8, 256, 0, stream>>>(tagLogits, RR, RR, targets,
                                                   srlProbs, accum, NROWS);
  ce_softmax_kernel<<<NROWS / 8, 256, 0, stream>>>(depOut, 50, 50, dep_tags,
                                                   nullptr, accum + 2, NROWS);
  finalize_kernel<<<1, 1, 0, stream>>>(accum, outScal);
}
